// Scaled_Dot_Product_Attention_16767552323972
// MI455X (gfx1250) — matrix operand order repaired and checked by reading, not compiled, not run on hardware
//
#include <hip/hip_runtime.h>
#include <math.h>

typedef __attribute__((ext_vector_type(16))) _Float16 v16h;
typedef __attribute__((ext_vector_type(8)))  _Float16 v8h;
typedef __attribute__((ext_vector_type(8)))  float    v8f;

#define QTILE 128
#define KTILE 32
#define DHEAD 128
#define LDK 136   // halves; 16 rows hit disjoint 4-bank groups for b128 reads
#define LDV 40    // Vt row stride (32 keys + 8 pad)
#define LDP 40

union V16 { v16h v; v8h h[2]; };

// convert 8 consecutive f32 -> v8h
static __device__ __forceinline__ v8h cvt8(const float* __restrict__ p) {
  float4 a = *(const float4*)p;
  float4 b = *(const float4*)(p + 4);
  v8h r;
  r[0] = (_Float16)a.x; r[1] = (_Float16)a.y; r[2] = (_Float16)a.z; r[3] = (_Float16)a.w;
  r[4] = (_Float16)b.x; r[5] = (_Float16)b.y; r[6] = (_Float16)b.z; r[7] = (_Float16)b.w;
  return r;
}

// store one staged float4 pair into K (row-major f16) and V (transposed f16) tiles
static __device__ __forceinline__ void store_tile(_Float16* __restrict__ ksb,
                                                  _Float16* __restrict__ vtb,
                                                  int tid, int i,
                                                  const float4& kx, const float4& vx) {
  int f4  = tid + i * 256;
  int row = f4 >> 5;            // 32 float4 per 128-wide row
  int c4  = (f4 & 31) << 2;
  _Float16* d = &ksb[row * LDK + c4];
  d[0] = (_Float16)kx.x; d[1] = (_Float16)kx.y;
  d[2] = (_Float16)kx.z; d[3] = (_Float16)kx.w;
  vtb[(c4 + 0) * LDV + row] = (_Float16)vx.x;
  vtb[(c4 + 1) * LDV + row] = (_Float16)vx.y;
  vtb[(c4 + 2) * LDV + row] = (_Float16)vx.z;
  vtb[(c4 + 3) * LDV + row] = (_Float16)vx.w;
}

__global__ __launch_bounds__(256)
void fa_fwd_f16wmma(const float* __restrict__ Qg, const float* __restrict__ Kg,
                    const float* __restrict__ Vg, const int* __restrict__ VL,
                    float* __restrict__ Og, int B, int NQ, int NK) {
  __shared__ _Float16 Ks[2][KTILE * LDK];   // [buf][key][dim]
  __shared__ _Float16 Vt[2][DHEAD * LDV];   // [buf][dim][key]
  __shared__ _Float16 Ps[8 * 16 * LDP];     // per-wave P relayout scratch

  const int tid  = threadIdx.x;
  const int w    = tid >> 5;
  const int lane = tid & 31;
  const int hi   = (lane >> 4) & 1;
  const int ln   = lane & 15;

  const int qb = blockIdx.x * QTILE;
  const int b  = blockIdx.y;
  const int vlen = VL[b];

  const float* Qp = Qg + ((size_t)b * NQ + qb) * DHEAD;
  const float* Kp = Kg + (size_t)b * NK * DHEAD;
  const float* Vp = Vg + (size_t)b * NK * DHEAD;

  // ---- Q tile straight from global into A-operand registers ----
  // A layout, chunk c: lanes 0-15 hold K = 32c+0..7 (h0) and 32c+16..23 (h1);
  // lanes 16-31 hold K = 32c+8..15 and 32c+24..31.
  V16 qa[4];
  {
    const float* qrow = Qp + (size_t)(w * 16 + ln) * DHEAD;
    #pragma unroll
    for (int c = 0; c < 4; ++c) {
      int kb = 32 * c + (hi ? 8 : 0);
      qa[c].h[0] = cvt8(qrow + kb);
      qa[c].h[1] = cvt8(qrow + kb + 16);
    }
  }

  v8f Oacc[8];
  #pragma unroll
  for (int t = 0; t < 8; ++t) Oacc[t] = (v8f){};
  float mrow[8], lrow[8];
  #pragma unroll
  for (int r = 0; r < 8; ++r) { mrow[r] = -INFINITY; lrow[r] = 0.0f; }

  const float scale = 0.08838834764831845f;  // 1/sqrt(128)
  const int nkeff = (vlen < NK) ? vlen : NK;
  const int ktEnd = (nkeff + KTILE - 1) / KTILE;  // masked tails underflow to 0

  // ---- prologue: stage tile 0 into buffer 0 ----
  #pragma unroll
  for (int i = 0; i < (KTILE * DHEAD) / (4 * 256); ++i) {  // 4 iters
    int f4 = tid + i * 256;
    int row = f4 >> 5;
    int c4  = (f4 & 31) << 2;
    float4 kx = *(const float4*)(Kp + (size_t)row * DHEAD + c4);
    float4 vx = *(const float4*)(Vp + (size_t)row * DHEAD + c4);
    store_tile(Ks[0], Vt[0], tid, i, kx, vx);
  }

  for (int kt = 0; kt < ktEnd; ++kt) {
    const int kbase = kt * KTILE;
    const int cur = kt & 1, nxt = cur ^ 1;
    // tile kt visible to all waves; all waves done reading buffer `nxt` (iter kt-1)
    __syncthreads();

    // ---- issue next tile's global loads early (overlap with WMMA) ----
    float4 kreg[4], vreg[4];
    const bool havenext = (kt + 1 < ktEnd);
    if (havenext) {
      const size_t nb = (size_t)(kbase + KTILE) * DHEAD;
      #pragma unroll
      for (int i = 0; i < 4; ++i) {
        int f4 = tid + i * 256;
        int row = f4 >> 5;
        int c4  = (f4 & 31) << 2;
        kreg[i] = *(const float4*)(Kp + nb + (size_t)row * DHEAD + c4);
        vreg[i] = *(const float4*)(Vp + nb + (size_t)row * DHEAD + c4);
      }
    }

    // ---- S = Q * K^T : two 16x16 N-subtiles, K=128 over 4 WMMA each ----
    v8f S0 = (v8f){}, S1 = (v8f){};
    {
      const _Float16* kr0 = &Ks[cur][ln * LDK];
      const _Float16* kr1 = &Ks[cur][(ln + 16) * LDK];
      #pragma unroll
      for (int c = 0; c < 4; ++c) {
        int kb = 32 * c + (hi ? 8 : 0);  // B: lanes 0-15 = K 0..7 and 16..23 (same as A)
        V16 b0, b1;
        b0.h[0] = *(const v8h*)(kr0 + kb);
        b0.h[1] = *(const v8h*)(kr0 + kb + 16);
        b1.h[0] = *(const v8h*)(kr1 + kb);
        b1.h[1] = *(const v8h*)(kr1 + kb + 16);
        S0 = __builtin_amdgcn_wmma_f32_16x16x32_f16(false, qa[c].v, false, b0.v,
                                                    (short)0, S0, false, false);
        S1 = __builtin_amdgcn_wmma_f32_16x16x32_f16(false, qa[c].v, false, b1.v,
                                                    (short)0, S1, false, false);
      }
    }

    // ---- scale + mask (column is lane-uniform in C layout) ----
    const float add0 = ((kbase + ln) < vlen) ? 0.0f : -INFINITY;
    const float add1 = ((kbase + 16 + ln) < vlen) ? 0.0f : -INFINITY;
    #pragma unroll
    for (int r = 0; r < 8; ++r) {
      S0[r] = S0[r] * scale + add0;
      S1[r] = S1[r] * scale + add1;
    }

    // ---- online softmax: butterfly over the 16-lane N-groups ----
    float p0[8], p1[8], corr[8];
    #pragma unroll
    for (int r = 0; r < 8; ++r) {
      float t = fmaxf(S0[r], S1[r]);
      t = fmaxf(t, __shfl_xor(t, 8, 32));
      t = fmaxf(t, __shfl_xor(t, 4, 32));
      t = fmaxf(t, __shfl_xor(t, 2, 32));
      t = fmaxf(t, __shfl_xor(t, 1, 32));
      float mnew = fmaxf(mrow[r], t);
      corr[r] = __expf(mrow[r] - mnew);
      mrow[r] = mnew;
      p0[r] = __expf(S0[r] - mnew);   // masked cols: exp(-inf)=0
      p1[r] = __expf(S1[r] - mnew);
      float s = p0[r] + p1[r];
      s += __shfl_xor(s, 8, 32);
      s += __shfl_xor(s, 4, 32);
      s += __shfl_xor(s, 2, 32);
      s += __shfl_xor(s, 1, 32);
      lrow[r] = lrow[r] * corr[r] + s;
    }
    #pragma unroll
    for (int t = 0; t < 8; ++t)
      #pragma unroll
      for (int r = 0; r < 8; ++r) Oacc[t][r] *= corr[r];

    // ---- re-layout P (C layout -> A layout) through per-wave LDS ----
    _Float16* pw = &Ps[w * 16 * LDP];
    #pragma unroll
    for (int r = 0; r < 8; ++r) {
      int row = r + hi * 8;
      pw[row * LDP + ln]      = (_Float16)p0[r];
      pw[row * LDP + 16 + ln] = (_Float16)p1[r];
    }
    asm volatile("s_wait_dscnt 0" ::: "memory");  // same-wave LDS RAW
    V16 pa;
    {
      const _Float16* pr = pw + ln * LDP;
      int kb = hi ? 8 : 0;
      pa.h[0] = *(const v8h*)(pr + kb);
      pa.h[1] = *(const v8h*)(pr + 16 + kb);
    }

    // ---- O += P * V : 8 N-tiles over d=128 ----
    #pragma unroll
    for (int t = 0; t < 8; ++t) {
      const _Float16* vr = &Vt[cur][(t * 16 + ln) * LDV];
      int kb = hi ? 8 : 0;
      V16 vb;
      vb.h[0] = *(const v8h*)(vr + kb);
      vb.h[1] = *(const v8h*)(vr + kb + 16);
      Oacc[t] = __builtin_amdgcn_wmma_f32_16x16x32_f16(false, pa.v, false, vb.v,
                                                       (short)0, Oacc[t], false, false);
    }

    // ---- drain staged registers into the other buffer ----
    if (havenext) {
      #pragma unroll
      for (int i = 0; i < 4; ++i)
        store_tile(Ks[nxt], Vt[nxt], tid, i, kreg[i], vreg[i]);
    }
  }

  // ---- normalize and store ----
  float inv[8];
  #pragma unroll
  for (int r = 0; r < 8; ++r) inv[r] = 1.0f / lrow[r];
  float* Op = Og + ((size_t)b * NQ + qb + w * 16) * DHEAD;
  #pragma unroll
  for (int t = 0; t < 8; ++t) {
    #pragma unroll
    for (int r = 0; r < 8; ++r) {
      int row = r + hi * 8;
      Op[(size_t)row * DHEAD + t * 16 + ln] = Oacc[t][r] * inv[r];
    }
  }
}

extern "C" void kernel_launch(void* const* d_in, const int* in_sizes, int n_in,
                              void* d_out, int out_size, void* d_ws, size_t ws_size,
                              hipStream_t stream) {
  (void)n_in; (void)out_size; (void)d_ws; (void)ws_size;
  const float* Q  = (const float*)d_in[0];
  const float* K  = (const float*)d_in[1];
  const float* V  = (const float*)d_in[2];
  const int*   VL = (const int*)d_in[3];
  float* O = (float*)d_out;

  const int B  = in_sizes[3];
  const int NQ = in_sizes[0] / (B * DHEAD);
  const int NK = in_sizes[1] / (B * DHEAD);

  dim3 grid(NQ / QTILE, B), block(256);
  fa_fwd_f16wmma<<<grid, block, 0, stream>>>(Q, K, V, VL, O, B, NQ, NK);
}
